// AtomPosGNN_4810363372609
// MI455X (gfx1250) — matrix operand order repaired and checked by reading, not compiled, not run on hardware
//
#include <hip/hip_runtime.h>

// ---------------------------------------------------------------------------
// Types for CDNA5 WMMA (wave32, v_wmma_f32_16x16x32_bf16)
// ---------------------------------------------------------------------------
typedef __bf16 bf16_t;
typedef bf16_t v16bf __attribute__((ext_vector_type(16)));
typedef float  v8f   __attribute__((ext_vector_type(8)));

union Frag16 {
    v16bf v;
    uint4 q[2];
};

#define NNODE 8192

__device__ __forceinline__ unsigned short f2b(float x) {
    // round-to-nearest-even f32 -> bf16 (finite inputs)
    unsigned u = __float_as_uint(x);
    unsigned r = (u + 0x7FFFu + ((u >> 16) & 1u)) >> 16;
    return (unsigned short)r;
}

// Stable softplus on hardware transcendentals (v_exp_f32 / v_log_f32):
// softplus(x) = max(x,0) + log(1 + exp(-|x|))
__device__ __forceinline__ float softplus_f(float x) {
    return fmaxf(x, 0.0f) + __logf(1.0f + __expf(-fabsf(x)));
}

// ---------------------------------------------------------------------------
// Fused: Ab_bf16[row][col] = (adj!=0 && col!=row) ? 1.0bf16 : 0   (exact)
//        norm[row] = rsqrt(max(#nonzeros(row) - 1, 1))
// ---------------------------------------------------------------------------
__global__ __launch_bounds__(256) void k_cvt_norm(const float* __restrict__ adj,
                                                  unsigned short* __restrict__ adjb,
                                                  float* __restrict__ norm) {
    const int row = blockIdx.x;
    const int tid = threadIdx.x;
    const float4* p = (const float4*)(adj + (size_t)row * NNODE);
    uint2* q = (uint2*)(adjb + (size_t)row * NNODE);
    float cnt = 0.0f;
    for (int j = tid; j < NNODE / 4; j += 256) {
        float4 v = p[j];
        const int col = j * 4;
        int c0 = (v.x != 0.0f), c1 = (v.y != 0.0f);
        int c2 = (v.z != 0.0f), c3 = (v.w != 0.0f);
        cnt += (float)(c0 + c1 + c2 + c3);
        unsigned h0 = (c0 && (col + 0) != row) ? 0x3F80u : 0u;
        unsigned h1 = (c1 && (col + 1) != row) ? 0x3F80u : 0u;
        unsigned h2 = (c2 && (col + 2) != row) ? 0x3F80u : 0u;
        unsigned h3 = (c3 && (col + 3) != row) ? 0x3F80u : 0u;
        uint2 pk;
        pk.x = h0 | (h1 << 16);
        pk.y = h2 | (h3 << 16);
        q[j] = pk;
    }
    __shared__ float red[256];
    red[tid] = cnt;
    __syncthreads();
    for (int s = 128; s > 0; s >>= 1) {
        if (tid < s) red[tid] += red[tid + s];
        __syncthreads();
    }
    if (tid == 0) norm[row] = rsqrtf(fmaxf(red[0] - 1.0f, 1.0f));
}

// Fallback (no room in ws for bf16 adjacency): norm only.
__global__ __launch_bounds__(256) void k_norm(const float* __restrict__ adj,
                                              float* __restrict__ norm) {
    const int row = blockIdx.x;
    const int tid = threadIdx.x;
    const float4* p = (const float4*)(adj + (size_t)row * NNODE);
    float cnt = 0.0f;
    for (int j = tid; j < NNODE / 4; j += 256) {
        float4 v = p[j];
        cnt += (float)(v.x != 0.0f) + (float)(v.y != 0.0f) +
               (float)(v.z != 0.0f) + (float)(v.w != 0.0f);
    }
    __shared__ float red[256];
    red[tid] = cnt;
    __syncthreads();
    for (int s = 128; s > 0; s >>= 1) {
        if (tid < s) red[tid] += red[tid + s];
        __syncthreads();
    }
    if (tid == 0) norm[row] = rsqrtf(fmaxf(red[0] - 1.0f, 1.0f));
}

// feat0[i][k] = concat(atom_emb, atom_pos), IN = 125 + 3 = 128
__global__ __launch_bounds__(256) void k_concat(const float* __restrict__ emb,
                                                const float* __restrict__ pos,
                                                float* __restrict__ feat0) {
    int idx = blockIdx.x * 256 + threadIdx.x;   // 8192*128 total
    int i = idx >> 7, k = idx & 127;
    feat0[idx] = (k < 125) ? emb[i * 125 + k] : pos[i * 3 + (k - 125)];
}

// WT[o][k] = bf16(W[k][o])  -- weights transposed for the WMMA B operand
__global__ __launch_bounds__(256) void k_wT(const float* __restrict__ W,
                                            unsigned short* __restrict__ WT,
                                            int Kin) {
    int idx = blockIdx.x * 256 + threadIdx.x;
    if (idx >= 512 * Kin) return;
    int o = idx / Kin, k = idx - o * Kin;
    WT[idx] = f2b(W[k * 512 + o]);
}

// XsT[k][i] = bf16(feat[i][k] * norm[i])  -- scaled features, transposed
__global__ __launch_bounds__(256) void k_scaleT(const float* __restrict__ feat,
                                                const float* __restrict__ norm,
                                                unsigned short* __restrict__ XsT,
                                                int Kin) {
    const int i0 = blockIdx.x * 32;
    const int k0 = blockIdx.y * 32;
    __shared__ float tile[32][33];
    const int t = threadIdx.x;
    {
        int il = t >> 3, kg = (t & 7) * 4;
        float4 v = *(const float4*)(feat + (size_t)(i0 + il) * Kin + k0 + kg);
        float nm = norm[i0 + il];
        tile[il][kg + 0] = v.x * nm;
        tile[il][kg + 1] = v.y * nm;
        tile[il][kg + 2] = v.z * nm;
        tile[il][kg + 3] = v.w * nm;
    }
    __syncthreads();
    {
        int kl = t >> 3, ig = (t & 7) * 4;
        unsigned short h0 = f2b(tile[ig + 0][kl]);
        unsigned short h1 = f2b(tile[ig + 1][kl]);
        unsigned short h2 = f2b(tile[ig + 2][kl]);
        unsigned short h3 = f2b(tile[ig + 3][kl]);
        uint2 pk;
        pk.x = (unsigned)h0 | ((unsigned)h1 << 16);
        pk.y = (unsigned)h2 | ((unsigned)h3 << 16);
        *(uint2*)(XsT + (size_t)(k0 + kl) * NNODE + i0 + ig) = pk;
    }
}

// ---------------------------------------------------------------------------
// Aggregation: Hs[i][n] = bf16( norm[i] * sum_j Ab[i][j] * XsT[n][j] )
// Block: 32 M-rows x (8 waves * WN*16 cols) = full N width, K-loop over 8192.
// PRE=true : direct global bf16 fragment loads, explicit double-buffering so
//            WMMA overlaps in-flight loads (non-zero s_wait_loadcnt).
// PRE=false: f32 adjacency staged via LDS with compare-select bf16.
// ---------------------------------------------------------------------------
template <int WN, bool PRE>
__global__ __launch_bounds__(256) void k_agg(const float* __restrict__ adjf,
                                             const unsigned short* __restrict__ adjb,
                                             const float* __restrict__ norm,
                                             const unsigned short* __restrict__ XsT,
                                             unsigned short* __restrict__ Hs) {
    constexpr int N = 8 * WN * 16;
    const int m0 = blockIdx.x * 32;
    const int tid = threadIdx.x;
    const int wave = tid >> 5, lane = tid & 31;
    const int kh = lane >> 4;   // K-half selector (A/B fragment)
    const int ml = lane & 15;   // row within 16 (A) / col within 16 (B, C)
    __shared__ __align__(16) unsigned short As[32][40];  // used when !PRE

    const v8f vzero = {0, 0, 0, 0, 0, 0, 0, 0};
    v8f acc[2][WN];
#pragma unroll
    for (int mi = 0; mi < 2; ++mi)
#pragma unroll
        for (int ni = 0; ni < WN; ++ni) acc[mi][ni] = vzero;

    if constexpr (PRE) {
        // base pointers (K advances by +k elements)
        const unsigned short* apb[2];
#pragma unroll
        for (int mi = 0; mi < 2; ++mi)
            apb[mi] = adjb + (size_t)(m0 + mi * 16 + ml) * NNODE + (size_t)kh * 8;
        const unsigned short* bpb[WN];
#pragma unroll
        for (int ni = 0; ni < WN; ++ni) {
            const int n = wave * (WN * 16) + ni * 16 + ml;
            bpb[ni] = XsT + (size_t)n * NNODE + (size_t)kh * 8;
        }

        auto load_frags = [&](Frag16(&a)[2], Frag16(&b)[WN], int k) {
#pragma unroll
            for (int mi = 0; mi < 2; ++mi) {
                a[mi].q[0] = *(const uint4*)(apb[mi] + k);
                a[mi].q[1] = *(const uint4*)(apb[mi] + k + 16);
                __builtin_prefetch(apb[mi] + k + 1024, 0, 1);  // 2 KB ahead
            }
#pragma unroll
            for (int ni = 0; ni < WN; ++ni) {
                b[ni].q[0] = *(const uint4*)(bpb[ni] + k);
                b[ni].q[1] = *(const uint4*)(bpb[ni] + k + 16);
            }
        };
        auto do_mma = [&](Frag16(&a)[2], Frag16(&b)[WN]) {
#pragma unroll
            for (int ni = 0; ni < WN; ++ni)
#pragma unroll
                for (int mi = 0; mi < 2; ++mi)
                    acc[mi][ni] = __builtin_amdgcn_wmma_f32_16x16x32_bf16(
                        false, a[mi].v, false, b[ni].v, (short)0, acc[mi][ni],
                        false, false);
        };

        Frag16 a0[2], b0[WN], a1[2], b1[WN];
        load_frags(a0, b0, 0);
        for (int k = 0; k < NNODE; k += 64) {
            load_frags(a1, b1, k + 32);
            do_mma(a0, b0);
            if (k + 64 < NNODE) load_frags(a0, b0, k + 64);
            do_mma(a1, b1);
        }
    } else {
        const int ar = tid >> 3;        // staging row 0..31
        const int ac = (tid & 7) * 4;   // staging col group
        const int grow = m0 + ar;
        for (int k = 0; k < NNODE; k += 32) {
            float4 av = *(const float4*)(adjf + (size_t)grow * NNODE + k + ac);
            // entries are exactly 0/1; drop self loop; no rounding needed
            unsigned h0 = (av.x != 0.0f && k + ac + 0 != grow) ? 0x3F80u : 0u;
            unsigned h1 = (av.y != 0.0f && k + ac + 1 != grow) ? 0x3F80u : 0u;
            unsigned h2 = (av.z != 0.0f && k + ac + 2 != grow) ? 0x3F80u : 0u;
            unsigned h3 = (av.w != 0.0f && k + ac + 3 != grow) ? 0x3F80u : 0u;
            uint2 pk;
            pk.x = h0 | (h1 << 16);
            pk.y = h2 | (h3 << 16);
            __syncthreads();
            *(uint2*)&As[ar][ac] = pk;
            __syncthreads();

            Frag16 a[2];
#pragma unroll
            for (int mi = 0; mi < 2; ++mi) {
                const unsigned short* ap = &As[mi * 16 + ml][0];
                a[mi].q[0] = *(const uint4*)(ap + kh * 8);
                a[mi].q[1] = *(const uint4*)(ap + 16 + kh * 8);
            }
#pragma unroll
            for (int ni = 0; ni < WN; ++ni) {
                const int n = wave * (WN * 16) + ni * 16 + ml;
                const unsigned short* bp = XsT + (size_t)n * NNODE + k + kh * 8;
                Frag16 b;
                b.q[0] = *(const uint4*)(bp);
                b.q[1] = *(const uint4*)(bp + 16);
#pragma unroll
                for (int mi = 0; mi < 2; ++mi)
                    acc[mi][ni] = __builtin_amdgcn_wmma_f32_16x16x32_bf16(
                        false, a[mi].v, false, b.v, (short)0, acc[mi][ni],
                        false, false);
            }
        }
    }

#pragma unroll
    for (int mi = 0; mi < 2; ++mi)
#pragma unroll
        for (int ni = 0; ni < WN; ++ni) {
            const int col = wave * (WN * 16) + ni * 16 + ml;
#pragma unroll
            for (int v = 0; v < 8; ++v) {
                const int row = m0 + mi * 16 + kh * 8 + v;
                Hs[(size_t)row * N + col] = f2b(acc[mi][ni][v] * norm[row]);
            }
        }
}

// ---------------------------------------------------------------------------
// Dense layer: out[i][o] = softplus( sum_k Hs[i][k]*WT[o][k] + bias[o] )
// Operands are L2-resident bf16; double-buffered fragment loads, no LDS.
// ---------------------------------------------------------------------------
template <int KIN>
__global__ __launch_bounds__(256) void k_layer(const unsigned short* __restrict__ Hs,
                                               const unsigned short* __restrict__ WT,
                                               const float* __restrict__ bias,
                                               float* __restrict__ out) {
    const int m0 = blockIdx.x * 32;
    const int tid = threadIdx.x;
    const int wave = tid >> 5, lane = tid & 31;
    const int kh = lane >> 4;
    const int ml = lane & 15;

    const v8f vzero = {0, 0, 0, 0, 0, 0, 0, 0};
    v8f acc[2][4];
#pragma unroll
    for (int mi = 0; mi < 2; ++mi)
#pragma unroll
        for (int ni = 0; ni < 4; ++ni) acc[mi][ni] = vzero;

    const unsigned short* apb[2];
#pragma unroll
    for (int mi = 0; mi < 2; ++mi)
        apb[mi] = Hs + (size_t)(m0 + mi * 16 + ml) * KIN + (size_t)kh * 8;
    const unsigned short* bpb[4];
#pragma unroll
    for (int ni = 0; ni < 4; ++ni) {
        const int n = wave * 64 + ni * 16 + ml;
        bpb[ni] = WT + (size_t)n * KIN + (size_t)kh * 8;
    }

    auto load_frags = [&](Frag16(&a)[2], Frag16(&b)[4], int k) {
#pragma unroll
        for (int mi = 0; mi < 2; ++mi) {
            a[mi].q[0] = *(const uint4*)(apb[mi] + k);
            a[mi].q[1] = *(const uint4*)(apb[mi] + k + 16);
        }
#pragma unroll
        for (int ni = 0; ni < 4; ++ni) {
            b[ni].q[0] = *(const uint4*)(bpb[ni] + k);
            b[ni].q[1] = *(const uint4*)(bpb[ni] + k + 16);
        }
    };
    auto do_mma = [&](Frag16(&a)[2], Frag16(&b)[4]) {
#pragma unroll
        for (int ni = 0; ni < 4; ++ni)
#pragma unroll
            for (int mi = 0; mi < 2; ++mi)
                acc[mi][ni] = __builtin_amdgcn_wmma_f32_16x16x32_bf16(
                    false, a[mi].v, false, b[ni].v, (short)0, acc[mi][ni],
                    false, false);
    };

    Frag16 a0[2], b0[4], a1[2], b1[4];
    load_frags(a0, b0, 0);
    for (int k = 0; k < KIN; k += 64) {
        load_frags(a1, b1, k + 32);
        do_mma(a0, b0);
        if (k + 64 < KIN) load_frags(a0, b0, k + 64);
        do_mma(a1, b1);
    }

#pragma unroll
    for (int ni = 0; ni < 4; ++ni) {
        const int col = wave * 64 + ni * 16 + ml;
        const float bc = bias[col];
#pragma unroll
        for (int mi = 0; mi < 2; ++mi)
#pragma unroll
            for (int v = 0; v < 8; ++v) {
                const int row = m0 + mi * 16 + kh * 8 + v;
                out[(size_t)row * 512 + col] = softplus_f(acc[mi][ni][v] + bc);
            }
    }
}

// ---------------------------------------------------------------------------
extern "C" void kernel_launch(void* const* d_in, const int* in_sizes, int n_in,
                              void* d_out, int out_size, void* d_ws, size_t ws_size,
                              hipStream_t stream) {
    (void)in_sizes; (void)n_in; (void)out_size;
    const float* pos = (const float*)d_in[0];
    const float* adj = (const float*)d_in[1];
    const float* emb = (const float*)d_in[2];
    const float* W[4] = {(const float*)d_in[3], (const float*)d_in[5],
                         (const float*)d_in[7], (const float*)d_in[9]};
    const float* B[4] = {(const float*)d_in[4], (const float*)d_in[6],
                         (const float*)d_in[8], (const float*)d_in[10]};

    char* w = (char*)d_ws;
    auto take = [&](size_t bytes) {
        char* p = w;
        w += (bytes + 255) & ~(size_t)255;
        return p;
    };
    float* norm  = (float*)take((size_t)NNODE * 4);
    float* feat0 = (float*)take((size_t)NNODE * 128 * 4);
    float* featA = (float*)take((size_t)NNODE * 512 * 4);
    unsigned short* XsT = (unsigned short*)take((size_t)512 * NNODE * 2);
    unsigned short* Hs  = (unsigned short*)take((size_t)NNODE * 512 * 2);
    unsigned short* WT[4];
    WT[0] = (unsigned short*)take((size_t)512 * 128 * 2);
    WT[1] = (unsigned short*)take((size_t)512 * 512 * 2);
    WT[2] = (unsigned short*)take((size_t)512 * 512 * 2);
    WT[3] = (unsigned short*)take((size_t)512 * 512 * 2);
    // bf16 adjacency last so the fallback layout is a prefix
    unsigned short* Ab = (unsigned short*)take((size_t)NNODE * NNODE * 2);
    const bool pre = ws_size >= (size_t)(w - (char*)d_ws);

    if (pre)
        k_cvt_norm<<<NNODE, 256, 0, stream>>>(adj, Ab, norm);
    else
        k_norm<<<NNODE, 256, 0, stream>>>(adj, norm);
    k_concat<<<(NNODE * 128) / 256, 256, 0, stream>>>(emb, pos, feat0);
    k_wT<<<(512 * 128 + 255) / 256, 256, 0, stream>>>(W[0], WT[0], 128);
    for (int l = 1; l < 4; ++l)
        k_wT<<<(512 * 512 + 255) / 256, 256, 0, stream>>>(W[l], WT[l], 512);

    // Layer 0 (input width 128)
    k_scaleT<<<dim3(NNODE / 32, 128 / 32), 256, 0, stream>>>(feat0, norm, XsT, 128);
    if (pre)
        k_agg<1, true><<<NNODE / 32, 256, 0, stream>>>(adj, Ab, norm, XsT, Hs);
    else
        k_agg<1, false><<<NNODE / 32, 256, 0, stream>>>(adj, Ab, norm, XsT, Hs);
    k_layer<128><<<NNODE / 32, 256, 0, stream>>>(Hs, WT[0], B[0], featA);

    // Layers 1..3 (width 512)
    for (int l = 1; l < 4; ++l) {
        k_scaleT<<<dim3(NNODE / 32, 512 / 32), 256, 0, stream>>>(featA, norm, XsT, 512);
        if (pre)
            k_agg<4, true><<<NNODE / 32, 256, 0, stream>>>(adj, Ab, norm, XsT, Hs);
        else
            k_agg<4, false><<<NNODE / 32, 256, 0, stream>>>(adj, Ab, norm, XsT, Hs);
        float* dst = (l == 3) ? (float*)d_out : featA;
        k_layer<512><<<NNODE / 32, 256, 0, stream>>>(Hs, WT[l], B[l], dst);
    }
}
